// SimpleLSTM_1657857376429
// MI455X (gfx1250) — matrix operand order repaired and checked by reading, not compiled, not run on hardware
//
#include <hip/hip_runtime.h>

// ---------------------------------------------------------------------------
// SimpleLSTM on MI455X (gfx1250): bf16 WMMA GEMMs + persistent grid-synced scan
//   B=32, T=1024, D=512, H=512, 4H=2048
// Data movement: async-to-LDS (ASYNCcnt) for h, TDM tensor_load_to_lds
// (TENSORcnt) double-buffered prefetch for zx tiles, WMMA for all GEMMs.
// ---------------------------------------------------------------------------

#define Bdim   32
#define Tdim   1024
#define Ddim   512
#define Hdim   512
#define FOURH  2048
#define NWG    32      // persistent workgroups for the scan kernel

typedef __attribute__((ext_vector_type(16))) __bf16 v16bf;
typedef __attribute__((ext_vector_type(8)))  __bf16 v8bf;
typedef __attribute__((ext_vector_type(4)))  __bf16 v4bf;
typedef __attribute__((ext_vector_type(8)))  float  v8f;
typedef __attribute__((ext_vector_type(4)))  unsigned int u32x4;
typedef __attribute__((ext_vector_type(8)))  unsigned int u32x8;

// ---- WMMA fragment helpers (16x16x32 bf16, wave32, contiguous k = 16h+s order, same in A and B) ----
// A (16x32): lanes 0-15 hold row M=lane, K = k0+0..15;
//            lanes 16-31 hold K = k0+16..31.
__device__ inline v16bf a_frag_lds(const __bf16* __restrict__ row, int k0, int lhalf) {
  const v8bf* p = (const v8bf*)(row + k0 + lhalf * 16);
  v8bf lo = p[0];
  v8bf hi = p[1];          // +8 elements
  v16bf a;
#pragma unroll
  for (int j = 0; j < 8; ++j) { a[j] = lo[j]; a[j + 8] = hi[j]; }
  return a;
}

// B (32x16): lane n (0-15) holds B[k0+0..15][n]; lane n+16 holds B[k0+16..31][n].
// Source here: transposed LDS tile Bt[N][K] -> contiguous 16 values per lane.
__device__ inline v16bf b_frag_lds_t(const __bf16* __restrict__ colrow, int lhalf) {
  const v8bf* p = (const v8bf*)(colrow + lhalf * 16);
  v8bf lo = p[0];
  v8bf hi = p[1];
  v16bf b;
#pragma unroll
  for (int j = 0; j < 8; ++j) { b[j] = lo[j]; b[j + 8] = hi[j]; }
  return b;
}

// Strided B fragment straight from a row-major bf16 [K x N] matrix (used once
// at scan start to park Wh in VGPRs).
__device__ inline v16bf b_frag_from_bf16(const __bf16* __restrict__ Bm, int ldb,
                                         int k0, int col, int lhalf) {
  const __bf16* p = Bm + (size_t)(k0 + lhalf * 16) * ldb + col;
  v16bf b;
#pragma unroll
  for (int j = 0; j < 16; ++j) b[j] = p[(size_t)j * ldb];
  return b;
}

// ---------------------------------------------------------------------------
// TDM: issue a 2D tensor_load_to_lds. Descriptor per CDNA5 ISA ch.8:
//  group0: [1:0]=count=1, [63:32]=lds byte addr, [120:64]=global addr,
//          [127:126]=type=2.
//  group1: [17:16]=data_size(log2 bytes), [79:48]=tensor_dim0,
//          [111:80]=tensor_dim1, [127:112]=tile_dim0, [143:128]=tile_dim1,
//          [207:160]=tensor_dim0_stride (element units).
// Per-wave instruction (EXEC ignored); caller gates to one wave per WG and
// synchronizes consumption with s_wait_tensorcnt + block barrier.
// ---------------------------------------------------------------------------
__device__ inline void tdm_load_2d(unsigned lds_off, const void* gaddr,
                                   unsigned dsz_log2, unsigned tile_d0,
                                   unsigned tile_d1,
                                   unsigned long long d0_stride) {
  unsigned long long ga = (unsigned long long)gaddr;
  const unsigned td0 = 0x40000000u, td1 = 0x40000000u;   // huge => no OOB clip
  u32x4 g0;
  g0[0] = 1u;                                            // count=1, user D#
  g0[1] = lds_off;
  g0[2] = (unsigned)(ga & 0xFFFFFFFFu);
  g0[3] = (unsigned)((ga >> 32) & 0x01FFFFFFu) | (2u << 30);  // type=2
  u32x8 g1;
  g1[0] = (dsz_log2 & 3u) << 16;                         // wg_mask=0, data_size
  g1[1] = (td0 & 0xFFFFu) << 16;
  g1[2] = (td0 >> 16) | ((td1 & 0xFFFFu) << 16);
  g1[3] = (td1 >> 16) | ((tile_d0 & 0xFFFFu) << 16);
  g1[4] = tile_d1 & 0xFFFFu;                             // tile_dim2 = 0
  g1[5] = (unsigned)(d0_stride & 0xFFFFFFFFull);
  g1[6] = (unsigned)((d0_stride >> 32) & 0xFFFFull);     // dim1_stride = 0
  g1[7] = 0u;
  asm volatile("tensor_load_to_lds %0, %1" :: "s"(g0), "s"(g1) : "memory");
}

// ---------------------------------------------------------------------------
// Kernel -1: convert Wi/Wh to bf16 copies in workspace (1M elements each).
// ---------------------------------------------------------------------------
__global__ __launch_bounds__(256) void lstm_wcvt_kernel(
    const float* __restrict__ Wi, const float* __restrict__ Wh,
    __bf16* __restrict__ Wib, __bf16* __restrict__ Whb) {
  size_t i = ((size_t)blockIdx.x * 256 + threadIdx.x) * 4;
  float4 a = *(const float4*)(Wi + i);
  float4 b = *(const float4*)(Wh + i);
  v4bf ab = { (__bf16)a.x, (__bf16)a.y, (__bf16)a.z, (__bf16)a.w };
  v4bf bb = { (__bf16)b.x, (__bf16)b.y, (__bf16)b.z, (__bf16)b.w };
  *(v4bf*)(Wib + i) = ab;
  *(v4bf*)(Whb + i) = bb;
}

// ---------------------------------------------------------------------------
// Kernel 0: init scan state (h0 -> bf16 buffer parity 0, c0 -> c buffer,
// zero the grid-barrier counters). Re-run every launch => deterministic.
// ---------------------------------------------------------------------------
__global__ __launch_bounds__(256) void lstm_init_kernel(
    const float* __restrict__ c0, const float* __restrict__ h0,
    float* __restrict__ cbuf, __bf16* __restrict__ hbuf,
    unsigned* __restrict__ bar, unsigned* __restrict__ gen) {
  int idx = blockIdx.x * blockDim.x + threadIdx.x;
  if (idx < Bdim * Hdim) {
    cbuf[idx] = c0[idx];
    hbuf[idx] = (__bf16)h0[idx];   // parity-0 buffer
  }
  if (idx == 0) { *bar = 0u; *gen = 0u; }
}

// ---------------------------------------------------------------------------
// Kernel 1: zx = x @ Wi   (M=32768, K=512, N=2048), bf16 WMMA, fp32 accum.
// Block = 128(M) x 64(N); K chunk = 32. x chunk converted to bf16 and staged
// in LDS; Wi chunk (already bf16) staged TRANSPOSED in LDS so B fragments are
// contiguous per lane. Each wave: one 16-row strip, 4 N-tiles, A reused 4x.
// ---------------------------------------------------------------------------
__global__ __launch_bounds__(256) void lstm_zx_gemm_kernel(
    const float* __restrict__ x, const __bf16* __restrict__ Wib,
    float* __restrict__ zx) {
  __shared__ __bf16 As[128][32];   // 8 KB, [M][K]
  __shared__ __bf16 Bt[64][32];    // 4 KB, [N][K] (transposed)

  const int tid   = threadIdx.x;
  const int lane  = tid & 31;
  const int wave  = tid >> 5;
  const int lhalf = lane >> 4;
  const int n16   = lane & 15;
  const int ncol0 = blockIdx.x * 64;
  const int mrow0 = blockIdx.y * 128;
  const int mrow  = mrow0 + wave * 16;

  v8f acc0 = {}, acc1 = {}, acc2 = {}, acc3 = {};

  const int ar = tid >> 1;               // A stage: row 0..127
  const int ac = (tid & 1) * 16;         //          col 0 or 16
  const int br = tid >> 3;               // B stage: k row 0..31
  const int bc = (tid & 7) * 8;          //          n col 0..56

  for (int kc = 0; kc < Ddim; kc += 32) {
    // --- stage A chunk: 128x32 fp32 -> bf16, coalesced b128 global loads ---
    {
      const float4* src = (const float4*)(x + (size_t)(mrow0 + ar) * Ddim + kc + ac);
      float4 f0 = src[0], f1 = src[1], f2 = src[2], f3 = src[3];
      v8bf o0 = { (__bf16)f0.x, (__bf16)f0.y, (__bf16)f0.z, (__bf16)f0.w,
                  (__bf16)f1.x, (__bf16)f1.y, (__bf16)f1.z, (__bf16)f1.w };
      v8bf o1 = { (__bf16)f2.x, (__bf16)f2.y, (__bf16)f2.z, (__bf16)f2.w,
                  (__bf16)f3.x, (__bf16)f3.y, (__bf16)f3.z, (__bf16)f3.w };
      *(v8bf*)&As[ar][ac]     = o0;
      *(v8bf*)&As[ar][ac + 8] = o1;
    }
    // --- stage B chunk: 32x64 bf16, coalesced 16B loads, transposed store ---
    {
      v8bf w = *(const v8bf*)(Wib + (size_t)(kc + br) * FOURH + ncol0 + bc);
#pragma unroll
      for (int j = 0; j < 8; ++j) Bt[bc + j][br] = w[j];
    }
    __syncthreads();

    v16bf a  = a_frag_lds(&As[wave * 16 + n16][0], 0, lhalf);
    v16bf b0 = b_frag_lds_t(&Bt[ 0 + n16][0], lhalf);
    v16bf b1 = b_frag_lds_t(&Bt[16 + n16][0], lhalf);
    v16bf b2 = b_frag_lds_t(&Bt[32 + n16][0], lhalf);
    v16bf b3 = b_frag_lds_t(&Bt[48 + n16][0], lhalf);
    acc0 = __builtin_amdgcn_wmma_f32_16x16x32_bf16(false, a, false, b0, (short)0, acc0, false, false);
    acc1 = __builtin_amdgcn_wmma_f32_16x16x32_bf16(false, a, false, b1, (short)0, acc1, false, false);
    acc2 = __builtin_amdgcn_wmma_f32_16x16x32_bf16(false, a, false, b2, (short)0, acc2, false, false);
    acc3 = __builtin_amdgcn_wmma_f32_16x16x32_bf16(false, a, false, b3, (short)0, acc3, false, false);
    __syncthreads();
  }

  // C layout: acc[r] holds (M = mrow + r + 8*lhalf, N = ncol + n16)
#pragma unroll
  for (int r = 0; r < 8; ++r) {
    float* po = zx + (size_t)(mrow + r + 8 * lhalf) * FOURH + ncol0 + n16;
    po[0]  = acc0[r];
    po[16] = acc1[r];
    po[32] = acc2[r];
    po[48] = acc3[r];
  }
}

// ---------------------------------------------------------------------------
// Grid-wide sense barrier (NWG persistent blocks), device-scope atomics.
// ---------------------------------------------------------------------------
__device__ inline void grid_sync(unsigned* __restrict__ bar, unsigned* __restrict__ gen) {
  __syncthreads();
  if (threadIdx.x == 0) {
    unsigned g = __hip_atomic_load(gen, __ATOMIC_ACQUIRE, __HIP_MEMORY_SCOPE_AGENT);
    unsigned a = __hip_atomic_fetch_add(bar, 1u, __ATOMIC_ACQ_REL, __HIP_MEMORY_SCOPE_AGENT);
    if (a == NWG - 1) {
      __hip_atomic_store(bar, 0u, __ATOMIC_RELAXED, __HIP_MEMORY_SCOPE_AGENT);
      __hip_atomic_fetch_add(gen, 1u, __ATOMIC_ACQ_REL, __HIP_MEMORY_SCOPE_AGENT);
    } else {
      while (__hip_atomic_load(gen, __ATOMIC_ACQUIRE, __HIP_MEMORY_SCOPE_AGENT) == g) {
        __builtin_amdgcn_s_sleep(2);
      }
    }
  }
  __syncthreads();
}

// ---------------------------------------------------------------------------
// Kernel 2: persistent LSTM scan. 32 WGs x 8 waves = 256 waves; each wave owns
// one 16x16 tile of z=[32,2048]. Its 512x16 slice of Wh (bf16) lives in 128
// VGPRs for the whole scan. Per step:
//   - h rows async-staged into LDS (ASYNCcnt)
//   - zx tile for step t+1 TDM-prefetched into double-buffered LDS (TENSORcnt)
//   - 16 x v_wmma_f32_16x16x32_bf16, publish z, grid-sync
//   - fp32 gates -> y / h(bf16) / c, grid-sync.
// LDS map (single block => base offset 0):
//   [0, 16384)        hs   : 16 rows x 512 bf16
//   [16384, 24576)    zxs0 : 16 x 128 fp32 (t even)
//   [24576, 32768)    zxs1 : 16 x 128 fp32 (t odd)
// ---------------------------------------------------------------------------
#define HS_BYTES   16384
#define ZXS_BYTES  8192

__global__ __launch_bounds__(256) void lstm_scan_kernel(
    const float* __restrict__ zx, const __bf16* __restrict__ Whb,
    const float* __restrict__ bias,
    float* __restrict__ zbuf, __bf16* __restrict__ hbuf,
    float* __restrict__ cbuf,
    float* __restrict__ y, float* __restrict__ cf, float* __restrict__ hf,
    unsigned* __restrict__ bar, unsigned* __restrict__ gen) {

  __shared__ __align__(16) char smem[HS_BYTES + 2 * ZXS_BYTES];
  __bf16* hs = (__bf16*)smem;

  const int tid   = threadIdx.x;
  const int lane  = tid & 31;
  const int wave  = tid >> 5;
  const int lhalf = lane >> 4;
  const int n16   = lane & 15;
  const int gw    = blockIdx.x * 8 + wave;   // 0..255 tiles
  const int mt    = gw >> 7;                 // 0..1   (batch tile; uniform per WG)
  const int nt    = gw & 127;                // 0..127 (gate-column tile)
  const int col   = nt * 16 + n16;
  const int gtid  = blockIdx.x * 256 + tid;  // 0..8191
  const int colbase = (blockIdx.x & 15) * 128;   // WG's 128-col zx slice

  // Resident Wh fragments: bf16 loads once, kept in 16 x v16bf = 128 VGPRs.
  v16bf whf[16];
#pragma unroll
  for (int c = 0; c < 16; ++c)
    whf[c] = b_frag_from_bf16(Whb, FOURH, 32 * c, col, lhalf);

  const float bias_c = bias[col];

  // TDM prologue: stage zx tile for t=0 into buffer 0 (wave 0 issues).
  if (wave == 0) {
    tdm_load_2d(HS_BYTES, zx + (size_t)mt * 16 * Tdim * FOURH + colbase,
                /*dsz=*/2u, /*tile_d0=*/128u, /*tile_d1=*/16u,
                /*d0_stride=*/(unsigned long long)Tdim * FOURH);
  }

  for (int t = 0; t < Tdim; ++t) {
    const int p = t & 1;
    const __bf16* hprev = hbuf + (size_t)p * (Bdim * Hdim);
    __bf16*       hnext = hbuf + (size_t)(p ^ 1) * (Bdim * Hdim);

    // Async-stage rows [mt*16, mt*16+16) of h into LDS: 1024 x 16B transfers,
    // 4 per thread, via the CDNA5 async-to-LDS engine (ASYNCcnt).
    {
      const __bf16* srcg = hprev + (size_t)mt * 16 * Hdim;
#pragma unroll
      for (int i = 0; i < 4; ++i) {
        unsigned      ldsoff = (unsigned)((tid + 256 * i) * 16);  // hs @ LDS 0
        const __bf16* g      = srcg + (size_t)(tid + 256 * i) * 8;
        asm volatile("global_load_async_to_lds_b128 %0, %1, off"
                     :: "v"(ldsoff), "v"(g) : "memory");
      }
      asm volatile("s_wait_asynccnt 0" ::: "memory");
    }
    if (wave == 0) __builtin_amdgcn_s_wait_tensorcnt(0);   // zx tile t ready
    __syncthreads();

    // Accumulator init from the TDM-staged zx tile (+bias).
    // acc[r] @ batch b = mt*16 + r + 8*lhalf, local col = wave*16 + n16.
    const float* zt = (const float*)(smem + HS_BYTES + p * ZXS_BYTES);
    v8f acc;
#pragma unroll
    for (int r = 0; r < 8; ++r)
      acc[r] = zt[(r + 8 * lhalf) * 128 + wave * 16 + n16] + bias_c;

    // Kick off TDM prefetch of next step's zx tile into the other buffer.
    // (That buffer was last read at t-1; two grid-syncs ago => safe to write.)
    if (t + 1 < Tdim && wave == 0) {
      tdm_load_2d(HS_BYTES + ((t + 1) & 1) * ZXS_BYTES,
                  zx + ((size_t)mt * 16 * Tdim + (t + 1)) * FOURH + colbase,
                  2u, 128u, 16u, (unsigned long long)Tdim * FOURH);
    }

    // K = 512 in 16 chunks of 32: 16 x v_wmma_f32_16x16x32_bf16
    const __bf16* arow = hs + (size_t)n16 * Hdim;
#pragma unroll
    for (int c = 0; c < 16; ++c) {
      v16bf a = a_frag_lds(arow, 32 * c, lhalf);
      acc = __builtin_amdgcn_wmma_f32_16x16x32_bf16(false, a, false, whf[c],
                                                    (short)0, acc, false, false);
    }

    // Publish z tile.
#pragma unroll
    for (int r = 0; r < 8; ++r) {
      int bb = mt * 16 + r + 8 * lhalf;
      zbuf[(size_t)bb * FOURH + col] = acc[r];
    }
    __threadfence();
    grid_sync(bar, gen);

    // Elementwise gates: 32*512 elements over 8192 threads, 2 consecutive each.
    {
      int base = gtid * 2;
      int bb   = base >> 9;
      const float* zr = zbuf + (size_t)bb * FOURH;
#pragma unroll
      for (int e = 0; e < 2; ++e) {
        int idx = base + e;
        int hh  = (idx & 511);
        float zi = zr[hh];
        float zf = zr[Hdim + hh];
        float zg = zr[2 * Hdim + hh];
        float zo = zr[3 * Hdim + hh];
        float ig = 1.0f / (1.0f + __expf(-zi));
        float fg = 1.0f / (1.0f + __expf(-zf));
        float gg = tanhf(zg);
        float og = 1.0f / (1.0f + __expf(-zo));
        float cc = cbuf[idx];
        float nc = fg * cc + ig * gg;
        float nh = og * tanhf(nc);
        cbuf[idx]  = nc;
        hnext[idx] = (__bf16)nh;
        y[((size_t)bb * Tdim + t) * Hdim + hh] = nh;
        if (t == Tdim - 1) { cf[idx] = nc; hf[idx] = nh; }
      }
    }
    __threadfence();
    grid_sync(bar, gen);
  }
}

// ---------------------------------------------------------------------------
// Launch
// ---------------------------------------------------------------------------
extern "C" void kernel_launch(void* const* d_in, const int* in_sizes, int n_in,
                              void* d_out, int out_size, void* d_ws, size_t ws_size,
                              hipStream_t stream) {
  (void)in_sizes; (void)n_in; (void)out_size; (void)ws_size;

  const float* x  = (const float*)d_in[0];   // [B,T,D]
  const float* c0 = (const float*)d_in[1];   // [B,H]
  const float* h0 = (const float*)d_in[2];   // [B,H]
  const float* Wi = (const float*)d_in[3];   // [D,4H]
  const float* Wh = (const float*)d_in[4];   // [H,4H]
  const float* bv = (const float*)d_in[5];   // [4H]

  float* out = (float*)d_out;
  float* cf = out;                            // [B,H]
  float* hf = out + Bdim * Hdim;              // [B,H]
  float* yo = out + 2 * Bdim * Hdim;          // [B,T,H]

  char* ws = (char*)d_ws;
  const size_t zx_bytes   = (size_t)Bdim * Tdim * FOURH * sizeof(float); // 256 MB
  const size_t zbuf_bytes = (size_t)Bdim * FOURH * sizeof(float);        // 256 KB
  const size_t hbuf_bytes = (size_t)2 * Bdim * Hdim * sizeof(__bf16);    //  64 KB
  const size_t cbuf_bytes = (size_t)Bdim * Hdim * sizeof(float);         //  64 KB
  const size_t wib_bytes  = (size_t)Ddim * FOURH * sizeof(__bf16);       //   2 MB
  const size_t whb_bytes  = (size_t)Hdim * FOURH * sizeof(__bf16);       //   2 MB

  size_t off = 0;
  float*    zx   = (float*)(ws + off);    off += zx_bytes;
  float*    zbuf = (float*)(ws + off);    off += zbuf_bytes;
  __bf16*   hbuf = (__bf16*)(ws + off);   off += hbuf_bytes;
  float*    cbuf = (float*)(ws + off);    off += cbuf_bytes;
  __bf16*   Wib  = (__bf16*)(ws + off);   off += wib_bytes;
  __bf16*   Whb  = (__bf16*)(ws + off);   off += whb_bytes;
  unsigned* bar  = (unsigned*)(ws + off);
  unsigned* gen  = bar + 16;

  // -1) weights -> bf16 copies (1M elems each, 4 per thread)
  lstm_wcvt_kernel<<<(Ddim * FOURH) / (256 * 4), 256, 0, stream>>>(Wi, Wh, Wib, Whb);

  // 0) init scan state + barrier counters (every call: deterministic)
  lstm_init_kernel<<<(Bdim * Hdim + 255) / 256, 256, 0, stream>>>(
      c0, h0, cbuf, hbuf, bar, gen);

  // 1) zx = x @ Wi : grid (N/64, M/128)
  lstm_zx_gemm_kernel<<<dim3(FOURH / 64, (Bdim * Tdim) / 128), 256, 0, stream>>>(
      x, Wib, zx);

  // 2) persistent recurrent scan
  lstm_scan_kernel<<<NWG, 256, 0, stream>>>(
      zx, Whb, bv, zbuf, hbuf, cbuf, yo, cf, hf, bar, gen);
}
